// SarvamMoEMLAAttention_68410239091006
// MI455X (gfx1250) — matrix operand order repaired and checked by reading, not compiled, not run on hardware
//
#include <hip/hip_runtime.h>
#include <hip/hip_bf16.h>
#include <math.h>

#define T 2048
#define HIDDEN 2048
#define NH 16
#define D_NOPE 128
#define D_ROPE 64
#define D_V 128
#define KV_LORA 512
#define D_QK (D_NOPE + D_ROPE)   // 192
#define D_KV (D_NOPE + D_V)      // 256
#define EPS 1e-6f
#define ROPE_THETA 10000.0f
#define NEG_BIG (-1.0e30f)
#define Q_SCALE 0.07216878364870322f   // (D_NOPE + D_ROPE)^-0.5, folded into Q cache

typedef __bf16 bf16_t;
typedef __attribute__((ext_vector_type(16))) __bf16        v16bf;
typedef __attribute__((ext_vector_type(4)))  __bf16        v4bf;
typedef __attribute__((ext_vector_type(8)))  float         v8f;
typedef __attribute__((ext_vector_type(4)))  float         f32x4;
typedef __attribute__((ext_vector_type(4)))  unsigned int  u32x4;

// Two contiguous 16B chunks -> one 16x32 bf16 WMMA fragment (wave32 layout):
//   A-frag lane(m,g): K in [g*8, g*8+8) then [16+g*8, 16+g*8+8)  (elements 0-7, 8-15)
//   B-frag lane(n,g): K in [g*8, g*8+8) then [16+g*8, 16+g*8+8)  (elements 0-7, 8-15)
union FragCvt { struct { u32x4 lo, hi; } u; v16bf v; };
union Pack8   { bf16_t b[8]; u32x4 q; };

__device__ __forceinline__ v8f wmma_bf16(v16bf a, v16bf b, v8f c) {
    return __builtin_amdgcn_wmma_f32_16x16x32_bf16(false, a, false, b,
                                                   (short)0, c, false, false);
}

// ISA 10.2: in the LDS aperture, flat_addr[31:0] == LDS byte offset, so the
// low 32 bits of a generic pointer to __shared__ are the async VDST operand.
__device__ __forceinline__ unsigned lds_off(const void* p) {
    return (unsigned)(size_t)p;
}
__device__ __forceinline__ void async_copy_b128(const void* gsrc, unsigned ldst) {
    asm volatile("global_load_async_to_lds_b128 %0, %1, off"
                 :: "v"(ldst), "v"(gsrc) : "memory");
}
__device__ __forceinline__ void wait_async0() {
    asm volatile("s_wait_asynccnt 0x0" ::: "memory");
}

// ---------------------------------------------------------------------------
// Elementwise f32 -> bf16 (row-major copy), 4 elems/thread.
// ---------------------------------------------------------------------------
__global__ __launch_bounds__(256) void to_bf16(const float* __restrict__ x,
                                               bf16_t* __restrict__ y, int n) {
    int i = (blockIdx.x * 256 + threadIdx.x) * 4;
    if (i >= n) return;
    f32x4 v = *(const f32x4*)(x + i);
    v4bf o;
    o[0] = (bf16_t)v[0]; o[1] = (bf16_t)v[1]; o[2] = (bf16_t)v[2]; o[3] = (bf16_t)v[3];
    *(v4bf*)(y + i) = o;
}

// ---------------------------------------------------------------------------
// W[K][N] f32  ->  Wt[N][K] bf16  (32x32 LDS tile transpose). K,N % 32 == 0.
// ---------------------------------------------------------------------------
__global__ __launch_bounds__(256) void transpose_to_bf16(const float* __restrict__ W,
                                                         bf16_t* __restrict__ Wt,
                                                         int K, int N) {
    __shared__ float tile[32][33];
    int n0 = blockIdx.x * 32, k0 = blockIdx.y * 32;
    int tx = threadIdx.x, ty = threadIdx.y;            // 32 x 8
#pragma unroll
    for (int j = 0; j < 4; ++j)
        tile[ty + j * 8][tx] = W[(size_t)(k0 + ty + j * 8) * N + n0 + tx];
    __syncthreads();
#pragma unroll
    for (int j = 0; j < 4; ++j)
        Wt[(size_t)(n0 + ty + j * 8) * K + k0 + tx] = (bf16_t)tile[tx][ty + j * 8];
}

// ---------------------------------------------------------------------------
// Blocked GEMM: C[M,N] = A[M,K] @ Bt[N,K]^T, bf16 operands, f32 accumulate.
// 256 threads = 8 waves; block tile 128x128; wave tile 32x64 (2x4 WMMA tiles).
// Double-buffered async global->LDS fills (ASYNCcnt) overlap tile k+1 DMA with
// tile k WMMAs. LDS rows padded to 80B so b128 fragment reads are conflict-free.
// Ragged N (576) is safe without zero-fill: N is 16-tile aligned, garbage B rows
// only feed tiles whose stores are skipped, and OOB reads stay inside workspace.
// MODE 0: C f32 row-major (col-bound checked).
// MODE 1: MLA kv epilogue -> Kout[t][h][192] (nope part, bf16) + Vt[h*128+vd][T].
// ---------------------------------------------------------------------------
#define LDSTRIDE 80
#define GBUF (128 * LDSTRIDE)
template <int MODE>
__global__ __launch_bounds__(256) void gemm_bt(const bf16_t* __restrict__ A,
                                               const bf16_t* __restrict__ Bt,
                                               float* __restrict__ C,
                                               bf16_t* __restrict__ Kout,
                                               bf16_t* __restrict__ Vt,
                                               int M, int N, int K) {
    __shared__ __align__(16) unsigned char smem[2 * 2 * GBUF];   // 2 buffers x (A|B)

    const int tid = threadIdx.x;
    const int lane = tid & 31, wid = tid >> 5;
    const int wr = wid & 3, wc = wid >> 2;             // 4 row-groups x 2 col-groups
    const int g = lane >> 4, ln = lane & 15;
    const int brow0 = blockIdx.y * 128;
    const int bcol0 = blockIdx.x * 128;

    const v8f vzero = {0.f, 0.f, 0.f, 0.f, 0.f, 0.f, 0.f, 0.f};
    v8f acc[2][4];
#pragma unroll
    for (int i = 0; i < 2; ++i)
#pragma unroll
        for (int j = 0; j < 4; ++j) acc[i][j] = vzero;

    // 512 x 16B chunks per matrix; each thread issues 2 A + 2 B async copies.
    auto issue_fill = [&](int buf, int k0) {
        unsigned char* sAb = smem + buf * (2 * GBUF);
        unsigned char* sBb = sAb + GBUF;
#pragma unroll
        for (int c = 0; c < 2; ++c) {
            int chunk = tid + c * 256;                 // 0..511
            int row = chunk >> 2, part = chunk & 3;
            async_copy_b128(A + (size_t)(brow0 + row) * K + k0 + part * 8,
                            lds_off(sAb + row * LDSTRIDE + part * 16));
            async_copy_b128(Bt + (size_t)(bcol0 + row) * K + k0 + part * 8,
                            lds_off(sBb + row * LDSTRIDE + part * 16));
        }
    };

    issue_fill(0, 0);
    int cur = 0;
    for (int k0 = 0; k0 < K; k0 += 32) {
        wait_async0();                                 // my writes for buffer `cur` done
        __syncthreads();                               // everyone's writes visible;
                                                       // all reads of other buffer done
        if (k0 + 32 < K) issue_fill(cur ^ 1, k0 + 32); // DMA next tile during math

        const unsigned char* sA = smem + cur * (2 * GBUF);
        const unsigned char* sB = sA + GBUF;

        FragCvt afr[2], bfr[4];
#pragma unroll
        for (int i = 0; i < 2; ++i) {
            const unsigned char* p = sA + (wr * 32 + i * 16 + ln) * LDSTRIDE;
            afr[i].u.lo = *(const u32x4*)(p + g * 16);
            afr[i].u.hi = *(const u32x4*)(p + 32 + g * 16);
        }
#pragma unroll
        for (int j = 0; j < 4; ++j) {
            const unsigned char* p = sB + (wc * 64 + j * 16 + ln) * LDSTRIDE;
            bfr[j].u.lo = *(const u32x4*)(p + g * 16);
            bfr[j].u.hi = *(const u32x4*)(p + 32 + g * 16);
        }
#pragma unroll
        for (int i = 0; i < 2; ++i)
#pragma unroll
            for (int j = 0; j < 4; ++j)
                acc[i][j] = wmma_bf16(afr[i].v, bfr[j].v, acc[i][j]);

        cur ^= 1;
    }

    // ---- epilogue ----
#pragma unroll
    for (int i = 0; i < 2; ++i) {
        int rbase = brow0 + wr * 32 + i * 16 + g * 8;  // token of acc[..][ii=0]
#pragma unroll
        for (int j = 0; j < 4; ++j) {
            int col = bcol0 + wc * 64 + j * 16 + ln;
            if (MODE == 0) {
                if (col < N) {
#pragma unroll
                    for (int ii = 0; ii < 8; ++ii)
                        C[(size_t)(rbase + ii) * N + col] = acc[i][j][ii];
                }
            } else {
                int h = col >> 8, hc = col & 255;      // 256 cols per head
                if (hc < D_NOPE) {                     // K-nope -> Kout[t][h][hc]
#pragma unroll
                    for (int ii = 0; ii < 8; ++ii)
                        Kout[((size_t)(rbase + ii) * NH + h) * D_QK + hc] =
                            (bf16_t)acc[i][j][ii];
                } else {                               // V -> Vt[h*128+vd][token]
                    int vd = h * D_V + (hc - D_NOPE);
                    Pack8 pk;
#pragma unroll
                    for (int ii = 0; ii < 8; ++ii) pk.b[ii] = (bf16_t)acc[i][j][ii];
                    *(u32x4*)(Vt + (size_t)vd * T + rbase) = pk.q;
                }
            }
        }
    }
}

// ---------------------------------------------------------------------------
// Per-token: RMSNorm(latent[:512]) -> kva_bf; RoPE(k_pe) broadcast into
// kbf[t][h][128..191]; Q cache qbf[t][h][192] = Q_SCALE*[q_nope | rope(q_pe)].
// ---------------------------------------------------------------------------
__global__ __launch_bounds__(256) void prep_kernel(const int* __restrict__ positions,
                                                   const float* __restrict__ latent,
                                                   const float* __restrict__ ln_w,
                                                   const float* __restrict__ qf,
                                                   bf16_t* __restrict__ kva_bf,
                                                   bf16_t* __restrict__ qbf,
                                                   bf16_t* __restrict__ kbf) {
    int t = blockIdx.x, tid = threadIdx.x;
    const float* lrow = latent + (size_t)t * (KV_LORA + D_ROPE);

    __shared__ float red[256];
    __shared__ float kpe_s[D_ROPE];
    float ss = 0.f;
    for (int i = tid; i < KV_LORA; i += 256) { float x = lrow[i]; ss += x * x; }
    red[tid] = ss; __syncthreads();
    for (int s = 128; s > 0; s >>= 1) {
        if (tid < s) red[tid] += red[tid + s];
        __syncthreads();
    }
    float inv = rsqrtf(red[0] * (1.0f / KV_LORA) + EPS);
    for (int i = tid; i < KV_LORA; i += 256)
        kva_bf[(size_t)t * KV_LORA + i] = (bf16_t)(lrow[i] * inv * ln_w[i]);

    float pos = (float)positions[t];
    if (tid < D_ROPE / 2) {
        float fr = pos * __powf(ROPE_THETA, -(2.0f * tid) / (float)D_ROPE);
        float c = __cosf(fr), s = __sinf(fr);
        float x1 = lrow[KV_LORA + 2 * tid], x2 = lrow[KV_LORA + 2 * tid + 1];
        kpe_s[2 * tid]     = x1 * c - x2 * s;
        kpe_s[2 * tid + 1] = x2 * c + x1 * s;
    }
    __syncthreads();

    for (int it = tid; it < NH * D_ROPE; it += 256) {
        int h = it >> 6, d = it & 63;
        kbf[((size_t)t * NH + h) * D_QK + D_NOPE + d] = (bf16_t)kpe_s[d];
    }
    const float* qr = qf + (size_t)t * (NH * D_QK);
    for (int it = tid; it < NH * D_NOPE; it += 256) {
        int h = it >> 7, d = it & 127;
        qbf[((size_t)t * NH + h) * D_QK + d] = (bf16_t)(qr[h * D_QK + d] * Q_SCALE);
    }
    for (int it = tid; it < NH * (D_ROPE / 2); it += 256) {
        int h = it >> 5, i = it & 31;
        float fr = pos * __powf(ROPE_THETA, -(2.0f * i) / (float)D_ROPE);
        float c = __cosf(fr), s = __sinf(fr);
        float x1 = qr[h * D_QK + D_NOPE + 2 * i], x2 = qr[h * D_QK + D_NOPE + 2 * i + 1];
        qbf[((size_t)t * NH + h) * D_QK + D_NOPE + 2 * i]     =
            (bf16_t)((x1 * c - x2 * s) * Q_SCALE);
        qbf[((size_t)t * NH + h) * D_QK + D_NOPE + 2 * i + 1] =
            (bf16_t)((x2 * c + x1 * s) * Q_SCALE);
    }
}

// ---------------------------------------------------------------------------
// Flash causal MLA attention, one wave per (16-query tile, head).
// All WMMA operands fed by pairs of b128 loads: qbf/kbf are [t][h][192] bf16,
// V is pre-transposed [h*128+vd][T] bf16. Fragment fetches are software-
// pipelined (2-slot double buffers) so each WMMA hides the next fetch's
// latency instead of stalling on s_wait_loadcnt 0.
// ---------------------------------------------------------------------------
__global__ __launch_bounds__(32) void mla_attn(const bf16_t* __restrict__ qbf,
                                               const bf16_t* __restrict__ kbf,
                                               const bf16_t* __restrict__ vt,
                                               bf16_t* __restrict__ attn_bf) {
    const int qt = blockIdx.x, h = blockIdx.y;
    const int lane = threadIdx.x;
    const int qbase = qt * 16;
    const int g = lane >> 4, ln = lane & 15;

    __shared__ float pb[16 * 36];                      // 36-word rows: conflict-free

    v16bf qa[6];
    {
        const unsigned char* qrow =
            (const unsigned char*)(qbf + ((size_t)(qbase + ln) * NH + h) * D_QK);
#pragma unroll
        for (int kb = 0; kb < 6; ++kb) {
            FragCvt f;
            f.u.lo = *(const u32x4*)(qrow + kb * 64 + g * 16);
            f.u.hi = *(const u32x4*)(qrow + kb * 64 + 32 + g * 16);
            qa[kb] = f.v;
        }
    }

    const v8f vzero = {0.f, 0.f, 0.f, 0.f, 0.f, 0.f, 0.f, 0.f};
    v8f acc[8];
#pragma unroll
    for (int c = 0; c < 8; ++c) acc[c] = vzero;
    float mrow[8], lrow_[8];
#pragma unroll
    for (int i = 0; i < 8; ++i) { mrow[i] = NEG_BIG; lrow_[i] = 0.f; }

    for (int kb0 = 0; kb0 < qbase + 16; kb0 += 32) {
        // ----- S[16x32] = Qs . K^T : 12 WMMAs, 2-slot pipelined K fetches -----
        v8f s0 = vzero, s1 = vzero;
        const unsigned char* kr0 =
            (const unsigned char*)(kbf + ((size_t)(kb0 + ln) * NH + h) * D_QK);
        const unsigned char* kr1 = kr0 + (size_t)16 * NH * D_QK * sizeof(bf16_t);

        FragCvt kA[2], kB[2];
        kA[0].u.lo = *(const u32x4*)(kr0 + g * 16);
        kA[0].u.hi = *(const u32x4*)(kr0 + 32 + g * 16);
        kB[0].u.lo = *(const u32x4*)(kr1 + g * 16);
        kB[0].u.hi = *(const u32x4*)(kr1 + 32 + g * 16);
#pragma unroll
        for (int kb = 0; kb < 6; ++kb) {
            int cs = kb & 1, ns = cs ^ 1;
            if (kb + 1 < 6) {
                kA[ns].u.lo = *(const u32x4*)(kr0 + (kb + 1) * 64 + g * 16);
                kA[ns].u.hi = *(const u32x4*)(kr0 + (kb + 1) * 64 + 32 + g * 16);
                kB[ns].u.lo = *(const u32x4*)(kr1 + (kb + 1) * 64 + g * 16);
                kB[ns].u.hi = *(const u32x4*)(kr1 + (kb + 1) * 64 + 32 + g * 16);
            }
            s0 = wmma_bf16(qa[kb], kA[cs].v, s0);
            s1 = wmma_bf16(qa[kb], kB[cs].v, s1);
        }

        // ----- preload first V fragment: softmax VALU hides its latency -----
        FragCvt vb[2];
        {
            const unsigned char* vr = (const unsigned char*)
                (vt + (size_t)(h * D_V + ln) * T + kb0 + g * 8);
            vb[0].u.lo = *(const u32x4*)vr;
            vb[0].u.hi = *(const u32x4*)(vr + 32);
        }

        // ----- causal mask + online softmax (scores pre-scaled via Q cache) -----
        __syncthreads();                               // pbuf reuse guard
#pragma unroll
        for (int i = 0; i < 8; ++i) {
            int r = i + 8 * g;
            int qidx = qbase + r;
            float v0 = s0[i]; if (kb0 + ln      > qidx) v0 = NEG_BIG;
            float v1 = s1[i]; if (kb0 + 16 + ln > qidx) v1 = NEG_BIG;

            float rm = fmaxf(v0, v1);
#pragma unroll
            for (int off = 8; off > 0; off >>= 1)
                rm = fmaxf(rm, __shfl_xor(rm, off, 32));

            float nm = fmaxf(mrow[i], rm);
            float rescale = __expf(mrow[i] - nm);
            float p0 = __expf(v0 - nm);
            float p1 = __expf(v1 - nm);

            float rs = p0 + p1;
#pragma unroll
            for (int off = 8; off > 0; off >>= 1)
                rs += __shfl_xor(rs, off, 32);

            lrow_[i] = lrow_[i] * rescale + rs;
            mrow[i]  = nm;
#pragma unroll
            for (int c = 0; c < 8; ++c) acc[c][i] *= rescale;

            pb[r * 36 + ln]      = p0;
            pb[r * 36 + 16 + ln] = p1;
        }
        __syncthreads();

        // ----- P (C layout, LDS) -> A fragment, b128 reads + cvt -----
        v16bf pf;
        {
            const float* prow = pb + ln * 36;
            f32x4 c0 = *(const f32x4*)(prow + g * 8);
            f32x4 c1 = *(const f32x4*)(prow + g * 8 + 4);
            f32x4 c2 = *(const f32x4*)(prow + 16 + g * 8);
            f32x4 c3 = *(const f32x4*)(prow + 16 + g * 8 + 4);
#pragma unroll
            for (int e = 0; e < 4; ++e) {
                pf[e]      = (bf16_t)c0[e];
                pf[e + 4]  = (bf16_t)c1[e];
                pf[e + 8]  = (bf16_t)c2[e];
                pf[e + 12] = (bf16_t)c3[e];
            }
        }

        // ----- acc += P(16x32) @ V(32x128), 2-slot pipelined V fetches -----
#pragma unroll
        for (int c = 0; c < 8; ++c) {
            int cs = c & 1, ns = cs ^ 1;
            if (c + 1 < 8) {
                const unsigned char* vr = (const unsigned char*)
                    (vt + (size_t)(h * D_V + (c + 1) * 16 + ln) * T + kb0 + g * 8);
                vb[ns].u.lo = *(const u32x4*)vr;
                vb[ns].u.hi = *(const u32x4*)(vr + 32);
            }
            acc[c] = wmma_bf16(pf, vb[cs].v, acc[c]);
        }
    }

    // ----- normalize, write bf16 attn cache [t][h*128+vd] -----
#pragma unroll
    for (int i = 0; i < 8; ++i) {
        int r = i + 8 * g;
        float invl = (lrow_[i] > 0.f) ? (1.0f / lrow_[i]) : 0.f;
#pragma unroll
        for (int c = 0; c < 8; ++c)
            attn_bf[(size_t)(qbase + r) * (NH * D_V) + h * D_V + c * 16 + ln] =
                (bf16_t)(acc[c][i] * invl);
    }
}

// ---------------------------------------------------------------------------
extern "C" void kernel_launch(void* const* d_in, const int* in_sizes, int n_in,
                              void* d_out, int out_size, void* d_ws, size_t ws_size,
                              hipStream_t stream) {
    const int*   positions = (const int*)  d_in[0];
    const float* hidden    = (const float*)d_in[1];
    const float* w_q       = (const float*)d_in[2];
    const float* w_kv_a    = (const float*)d_in[3];
    const float* kv_ln_w   = (const float*)d_in[4];
    const float* w_kv_b    = (const float*)d_in[5];
    const float* w_o       = (const float*)d_in[6];
    float* out = (float*)d_out;
    (void)in_sizes; (void)n_in; (void)out_size; (void)ws_size;

    unsigned char* wsp = (unsigned char*)d_ws;
    auto alloc = [&](size_t bytes) -> void* {
        void* p = (void*)wsp;
        wsp += (bytes + 255) & ~(size_t)255;
        return p;
    };
    bf16_t* hidden_bf = (bf16_t*)alloc((size_t)T * HIDDEN * 2);
    bf16_t* wq_t      = (bf16_t*)alloc((size_t)(NH * D_QK) * HIDDEN * 2);
    bf16_t* wkva_t    = (bf16_t*)alloc((size_t)(KV_LORA + D_ROPE) * HIDDEN * 2);
    bf16_t* wkvb_t    = (bf16_t*)alloc((size_t)(NH * D_KV) * KV_LORA * 2);
    bf16_t* wo_t      = (bf16_t*)alloc((size_t)HIDDEN * (NH * D_V) * 2);
    float*  qf32      = (float*) alloc((size_t)T * NH * D_QK * 4);
    float*  latent    = (float*) alloc((size_t)T * (KV_LORA + D_ROPE) * 4);
    bf16_t* kva_bf    = (bf16_t*)alloc((size_t)T * KV_LORA * 2);
    bf16_t* qbf       = (bf16_t*)alloc((size_t)T * NH * D_QK * 2);
    bf16_t* kbf       = (bf16_t*)alloc((size_t)T * NH * D_QK * 2);
    bf16_t* vt        = (bf16_t*)alloc((size_t)NH * D_V * T * 2);
    bf16_t* attn_bf   = (bf16_t*)alloc((size_t)T * NH * D_V * 2);

    dim3 tb(32, 8);
    to_bf16<<<(T * HIDDEN / 4 + 255) / 256, 256, 0, stream>>>(hidden, hidden_bf, T * HIDDEN);
    transpose_to_bf16<<<dim3((NH * D_QK) / 32, HIDDEN / 32), tb, 0, stream>>>(w_q, wq_t, HIDDEN, NH * D_QK);
    transpose_to_bf16<<<dim3((KV_LORA + D_ROPE) / 32, HIDDEN / 32), tb, 0, stream>>>(w_kv_a, wkva_t, HIDDEN, KV_LORA + D_ROPE);
    transpose_to_bf16<<<dim3((NH * D_KV) / 32, KV_LORA / 32), tb, 0, stream>>>(w_kv_b, wkvb_t, KV_LORA, NH * D_KV);
    transpose_to_bf16<<<dim3(HIDDEN / 32, HIDDEN / 32), tb, 0, stream>>>(w_o, wo_t, HIDDEN, NH * D_V);

    gemm_bt<0><<<dim3((NH * D_QK) / 128, T / 128), 256, 0, stream>>>(
        hidden_bf, wq_t, qf32, nullptr, nullptr, T, NH * D_QK, HIDDEN);
    gemm_bt<0><<<dim3((KV_LORA + D_ROPE + 127) / 128, T / 128), 256, 0, stream>>>(
        hidden_bf, wkva_t, latent, nullptr, nullptr, T, KV_LORA + D_ROPE, HIDDEN);
    prep_kernel<<<dim3(T), dim3(256), 0, stream>>>(
        positions, latent, kv_ln_w, qf32, kva_bf, qbf, kbf);
    gemm_bt<1><<<dim3((NH * D_KV) / 128, T / 128), 256, 0, stream>>>(
        kva_bf, wkvb_t, nullptr, kbf, vt, T, NH * D_KV, KV_LORA);
    mla_attn<<<dim3(T / 16, NH), dim3(32), 0, stream>>>(qbf, kbf, vt, attn_bf);
    gemm_bt<0><<<dim3(HIDDEN / 128, T / 128), 256, 0, stream>>>(
        attn_bf, wo_t, out, nullptr, nullptr, T, HIDDEN, HIDDEN);
}
